// Mamba2Mixer_46531675685257
// MI455X (gfx1250) — matrix operand order repaired and checked by reading, not compiled, not run on hardware
//
#include <hip/hip_runtime.h>

// ---------------- problem constants (from reference) ----------------
#define B_SZ     2
#define SEQ      2048
#define DMODEL   2048
#define INTER    4096
#define HEADS    64
#define PDIM     64
#define NSTATE   128
#define KCONV    4
#define CONV_DIM (INTER + 2 * NSTATE)   // 4352
#define EPS      1e-5f

#define GAS __attribute__((address_space(1)))
#define LAS __attribute__((address_space(3)))

// ---------------- WMMA types (CDNA5 / gfx1250, wave32) ----------------
typedef __attribute__((ext_vector_type(16))) __bf16 v16bf;
typedef __attribute__((ext_vector_type(8)))  float  v8f;
typedef int v4i __attribute__((vector_size(16)));   // matches async-LDS builtin param

union FragBF { v16bf v; unsigned int u[8]; };
union FragF  { v8f   v; float        f[8]; };

// ---------------- helpers ----------------
__device__ inline unsigned short f2bf(float f) {
  unsigned int u = __float_as_uint(f);
  u += 0x7FFFu + ((u >> 16) & 1u);        // round-to-nearest-even
  return (unsigned short)(u >> 16);
}

__device__ inline float silu_f(float x) {
  return x / (1.0f + __expf(-x));
}

// async copy of 16 bytes global -> LDS (no VGPR round trip, ASYNCcnt-tracked)
__device__ inline void async_b128(const unsigned short* gsrc, unsigned int* ldst) {
  __builtin_amdgcn_global_load_async_to_lds_b128(
      (GAS v4i*)((GAS void*)(void*)gsrc),
      (LAS v4i*)((LAS void*)(void*)ldst), 0, 0);
}

// ---------------- fp32 -> bf16 conversion (row-major, for activations) ----
__global__ __launch_bounds__(256)
void f32_to_bf16_kernel(const float* __restrict__ in,
                        unsigned short* __restrict__ out, long n) {
  long i = (long)blockIdx.x * 256 + threadIdx.x;
  if (i < n) out[i] = f2bf(in[i]);
}

// ---- fp32 -> bf16 transposed weight conversion: out[n][k] = bf16(in[k][n])
// out has Npad rows (rows n >= N are zero-filled). grid: (Npad/32, Kd/32).
__global__ __launch_bounds__(256)
void f32_to_bf16_transpose_kernel(const float* __restrict__ in,
                                  unsigned short* __restrict__ out,
                                  int Kd, int N, int Npad) {
  __shared__ unsigned short tile[32][33];
  const int tx = threadIdx.x & 31;
  const int ty = threadIdx.x >> 5;          // 0..7
  const int kBase = blockIdx.y * 32;
  const int nBase = blockIdx.x * 32;
#pragma unroll
  for (int j = 0; j < 4; ++j) {
    int kr = ty * 4 + j;
    int n  = nBase + tx;
    float v = (n < N) ? in[(long)(kBase + kr) * N + n] : 0.0f;
    tile[kr][tx] = f2bf(v);
  }
  __syncthreads();
#pragma unroll
  for (int j = 0; j < 4; ++j) {
    int r = ty * 4 + j;                     // n offset within tile
    out[(long)(nBase + r) * Kd + kBase + tx] = tile[tx][r];
  }
}

// ---------------- bf16 WMMA GEMM: C[M,N] = A[M,K] * Bt[N,K]^T ----------------
// A: [M][Kd] bf16 row-major. Bt: [N][Kd] bf16 (i.e. W pre-transposed, K-major).
// C: f32 row-major. Block: 256 threads (8 waves). Block tile 128(M) x 128(N),
// wave tile 32x64 (2x4 WMMAs). K stepped by 32, double-buffered LDS staged
// with GLOBAL_LOAD_ASYNC_TO_LDS_B128 (4 issues/thread/tile).
// Requires: M % 128 == 0, N % 128 == 0, K % 32 == 0.
__global__ __launch_bounds__(256)
void gemm_bf16_wmma_kernel(const unsigned short* __restrict__ A,
                           const unsigned short* __restrict__ Bt,
                           float* __restrict__ C,
                           int M, int Kd, int Ncols) {
  __shared__ unsigned int ldsA[2][128 * 16]; // [row][kpair] : 128 x 32 bf16
  __shared__ unsigned int ldsB[2][128 * 16]; // [col][kpair] : 128 x 32 bf16

  const int tid      = threadIdx.x;
  const int wave     = tid >> 5;
  const int lane     = tid & 31;
  const int laneHalf = lane >> 4;
  const int lane15   = lane & 15;
  const int waveM    = wave >> 1;   // 0..3 -> 32 rows each
  const int waveN    = wave & 1;    // 0..1 -> 64 cols each
  const int blockM   = blockIdx.y * 128;
  const int blockN   = blockIdx.x * 128;

  FragF acc[2][4];
#pragma unroll
  for (int i = 0; i < 2; ++i)
#pragma unroll
    for (int j = 0; j < 4; ++j)
#pragma unroll
      for (int r = 0; r < 8; ++r) acc[i][j].f[r] = 0.0f;

  // per-tile async staging: 128 rows x 64B = 512 x b128 chunks for each of A/B
  auto issue_tile = [&](int k0, int buf) {
#pragma unroll
    for (int i = 0; i < 2; ++i) {
      int idx = i * 256 + tid;              // 0..511
      int row = idx >> 2;
      int c4  = idx & 3;                    // 16B chunk within the 64B row slice
      async_b128(A  + (long)(blockM + row) * Kd + k0 + c4 * 8, &ldsA[buf][idx * 4]);
    }
#pragma unroll
    for (int i = 0; i < 2; ++i) {
      int idx = i * 256 + tid;
      int row = idx >> 2;
      int c4  = idx & 3;
      async_b128(Bt + (long)(blockN + row) * Kd + k0 + c4 * 8, &ldsB[buf][idx * 4]);
    }
  };

  const int nk = Kd >> 5;                   // K tiles of 32
  issue_tile(0, 0);

  for (int kt = 0; kt < nk; ++kt) {
    const int cur = kt & 1;
    if (kt + 1 < nk) {
      issue_tile((kt + 1) << 5, cur ^ 1);
      __builtin_amdgcn_s_wait_asynccnt(4);  // tile kt's 4 issues complete
    } else {
      __builtin_amdgcn_s_wait_asynccnt(0);
    }
    __syncthreads();                        // all waves' portions of tile kt landed

    // ---- build fragments per documented CDNA5 wave32 layouts ----
    FragBF aF[2], bF[4];
#pragma unroll
    for (int i = 0; i < 2; ++i) {
      int mrow = waveM * 32 + i * 16 + lane15;
#pragma unroll
      for (int v = 0; v < 8; ++v) {
        int pairIdx = (v < 4 ? v : v + 4) + laneHalf * 4;   // A 16x32 layout
        aF[i].u[v] = ldsA[cur][mrow * 16 + pairIdx];
      }
    }
#pragma unroll
    for (int j = 0; j < 4; ++j) {
      int col = waveN * 64 + j * 16 + lane15;
#pragma unroll
      for (int v = 0; v < 8; ++v) {
        int pairIdx = (v < 4 ? v : v + 4) + laneHalf * 4;   // B 32x16 layout
        bF[j].u[v] = ldsB[cur][col * 16 + pairIdx];
      }
    }

    // ---- 8 WMMAs per wave per K-step ----
#pragma unroll
    for (int i = 0; i < 2; ++i)
#pragma unroll
      for (int j = 0; j < 4; ++j)
        acc[i][j].v = __builtin_amdgcn_wmma_f32_16x16x32_bf16(
            false, aF[i].v, false, bF[j].v, (short)0, acc[i][j].v, false, false);

    __syncthreads();                        // done reading buf before it is refilled
  }

  // ---- store C (16x16 f32 D layout: VGPR r -> M = r + 8*laneHalf, N = lane15) ----
#pragma unroll
  for (int i = 0; i < 2; ++i) {
#pragma unroll
    for (int j = 0; j < 4; ++j) {
      int rbase = blockM + waveM * 32 + i * 16 + laneHalf * 8;
      int cbase = blockN + waveN * 64 + j * 16 + lane15;
#pragma unroll
      for (int r = 0; r < 8; ++r)
        C[(long)(rbase + r) * Ncols + cbase] = acc[i][j].f[r];
    }
  }
}

// ---------------- dt epilogue: compact + softplus(dt_raw + bias) ----------------
// raw: [rows][128] (padded GEMM out), out: [rows][64]
__global__ __launch_bounds__(256)
void dt_act_kernel(const float* __restrict__ raw, const float* __restrict__ dt_bias,
                   float* __restrict__ out, long total) {
  long i = (long)blockIdx.x * 256 + threadIdx.x;
  if (i >= total) return;
  long r = i >> 6;
  int  h = (int)(i & (HEADS - 1));
  float x = raw[r * 128 + h] + dt_bias[h];
  out[i] = (x < 10.0f) ? log1pf(__expf(fminf(x, 10.0f))) : x;
}

// ---------------- causal depthwise conv1d (K=4) + SiLU ----------------
__global__ __launch_bounds__(256)
void conv_silu_kernel(const float* __restrict__ xBC,
                      const float* __restrict__ cw,    // [CONV_DIM, 4]
                      const float* __restrict__ cb,    // [CONV_DIM]
                      float* __restrict__ out, long total) {
  long i = (long)blockIdx.x * 256 + threadIdx.x;
  if (i >= total) return;
  int  c  = (int)(i % CONV_DIM);
  long bl = i / CONV_DIM;
  int  l  = (int)(bl % SEQ);
  float acc = cb[c];
#pragma unroll
  for (int k = 0; k < KCONV; ++k) {
    int lk = l - (KCONV - 1) + k;
    float xv = (lk >= 0) ? xBC[i + (long)(k - (KCONV - 1)) * CONV_DIM] : 0.0f;
    acc = fmaf(xv, cw[c * KCONV + k], acc);
  }
  out[i] = silu_f(acc);
}

// ---------------- sequential SSM scan ----------------
// grid: (HEADS, B). 256 threads: p = tid>>2 (0..63), quadrant nq = tid&3 owns 32 n's.
// state kept in registers (32 f32/thread). B/C rows staged in LDS per timestep.
__global__ __launch_bounds__(256)
void ssm_scan_kernel(const float* __restrict__ xBC,   // activated [B, L, CONV_DIM]
                     const float* __restrict__ dt,    // [B, L, H]
                     const float* __restrict__ A_log, // [H]
                     const float* __restrict__ Dv,    // [H]
                     float* __restrict__ y) {         // [B, L, H, P]
  const int h   = blockIdx.x;
  const int b   = blockIdx.y;
  const int tid = threadIdx.x;
  const int p   = tid >> 2;
  const int nq  = tid & 3;

  const float Ah = -__expf(A_log[h]);
  const float Dh = Dv[h];

  float st[32];
#pragma unroll
  for (int j = 0; j < 32; ++j) st[j] = 0.0f;

  __shared__ float sB[NSTATE];
  __shared__ float sC[NSTATE];

  const float* base = xBC + (long)b * SEQ * CONV_DIM;

  for (int t = 0; t < SEQ; ++t) {
    const float* row = base + (long)t * CONV_DIM;
    if (tid < NSTATE)            sB[tid]          = row[INTER + tid];
    else                         sC[tid - NSTATE] = row[INTER + NSTATE + (tid - NSTATE)];
    __syncthreads();

    float dth = dt[((long)b * SEQ + t) * HEADS + h];
    float dA  = __expf(dth * Ah);
    float xp  = row[h * PDIM + p];
    float dtx = dth * xp;

    float yp = 0.0f;
#pragma unroll
    for (int j = 0; j < 32; ++j) {
      int n = nq * 32 + j;
      float s = fmaf(st[j], dA, dtx * sB[n]);
      st[j] = s;
      yp = fmaf(s, sC[n], yp);
    }
    // reduce the 4 quadrants (contiguous lanes within a wave32)
    yp += __shfl_xor(yp, 1);
    yp += __shfl_xor(yp, 2);
    if (nq == 0)
      y[(((long)b * SEQ + t) * HEADS + h) * PDIM + p] = yp + Dh * xp;
    __syncthreads();
  }
}

// ---------------- gated RMSNorm (G=1 -> over full INTER), bf16 out ----------------
__global__ __launch_bounds__(256)
void rmsnorm_gate_kernel(const float* __restrict__ y,
                         const float* __restrict__ gate,
                         const float* __restrict__ norm_w,
                         unsigned short* __restrict__ out) {
  long row = blockIdx.x;            // 0 .. B*SEQ-1
  int  tid = threadIdx.x;
  const float* yr = y    + row * INTER;
  const float* gr = gate + row * INTER;

  float v[16];
  float ss = 0.0f;
#pragma unroll
  for (int i = 0; i < 16; ++i) {
    int c = i * 256 + tid;
    float hv = yr[c] * silu_f(gr[c]);
    v[i] = hv;
    ss = fmaf(hv, hv, ss);
  }
  __shared__ float red[256];
  red[tid] = ss;
  __syncthreads();
  for (int s = 128; s > 0; s >>= 1) {
    if (tid < s) red[tid] += red[tid + s];
    __syncthreads();
  }
  float inv = rsqrtf(red[0] / (float)INTER + EPS);
#pragma unroll
  for (int i = 0; i < 16; ++i) {
    int c = i * 256 + tid;
    out[row * INTER + c] = f2bf(norm_w[c] * v[i] * inv);
  }
}

// ---------------- host orchestration ----------------
static inline size_t alignup(size_t x) { return (x + 255) & ~(size_t)255; }

extern "C" void kernel_launch(void* const* d_in, const int* in_sizes, int n_in,
                              void* d_out, int out_size, void* d_ws, size_t ws_size,
                              hipStream_t stream) {
  const float* hs      = (const float*)d_in[0];
  const float* W_z     = (const float*)d_in[1];
  const float* W_xBC   = (const float*)d_in[2];
  const float* W_dt    = (const float*)d_in[3];
  const float* conv_w  = (const float*)d_in[4];
  const float* conv_b  = (const float*)d_in[5];
  const float* dt_bias = (const float*)d_in[6];
  const float* A_log   = (const float*)d_in[7];
  const float* Dvec    = (const float*)d_in[8];
  const float* norm_w  = (const float*)d_in[9];
  const float* W_out   = (const float*)d_in[10];
  float* out = (float*)d_out;

  const long M        = (long)B_SZ * SEQ;        // 4096
  const int  DT_NPAD  = 128;
  const long n_hs     = M * DMODEL;
  const long n_Wz     = (long)DMODEL * INTER;
  const long n_WxBC   = (long)DMODEL * CONV_DIM;
  const long n_Wdt_t  = (long)DT_NPAD * DMODEL;
  const long n_Wout   = (long)INTER * DMODEL;
  const long n_gate   = M * INTER;
  const long n_xBC    = M * CONV_DIM;
  const long n_dtraw  = M * DT_NPAD;
  const long n_dt     = M * HEADS;

  // workspace layout
  char* p = (char*)d_ws;
  unsigned short* hs_bf    = (unsigned short*)p; p += alignup(n_hs    * 2);
  unsigned short* Wz_t     = (unsigned short*)p; p += alignup(n_Wz    * 2);
  unsigned short* WxBC_t   = (unsigned short*)p; p += alignup(n_WxBC  * 2);
  unsigned short* Wdt_t    = (unsigned short*)p; p += alignup(n_Wdt_t * 2);
  unsigned short* Wout_t   = (unsigned short*)p; p += alignup(n_Wout  * 2);
  float* gate    = (float*)p; p += alignup(n_gate  * 4);
  float* xBC_raw = (float*)p; p += alignup(n_xBC   * 4);
  float* xBC_act = (float*)p; p += alignup(n_xBC   * 4);
  float* dt_raw  = (float*)p; p += alignup(n_dtraw * 4);
  float* dtb     = (float*)p; p += alignup(n_dt    * 4);
  float* ybuf    = (float*)p; p += alignup(n_gate  * 4);
  unsigned short* normed_bf = (unsigned short*)p; p += alignup(n_gate * 2);

  // 1) activations: fp32 -> bf16 (row-major)
  f32_to_bf16_kernel<<<dim3((unsigned)((n_hs + 255) / 256)), dim3(256), 0, stream>>>(
      hs, hs_bf, n_hs);

  // 2) weights: fp32 -> bf16 transposed (K-major), dt padded to 128 cols
  f32_to_bf16_transpose_kernel<<<dim3(INTER / 32, DMODEL / 32), dim3(256), 0, stream>>>(
      W_z, Wz_t, DMODEL, INTER, INTER);
  f32_to_bf16_transpose_kernel<<<dim3(CONV_DIM / 32, DMODEL / 32), dim3(256), 0, stream>>>(
      W_xBC, WxBC_t, DMODEL, CONV_DIM, CONV_DIM);
  f32_to_bf16_transpose_kernel<<<dim3(DT_NPAD / 32, DMODEL / 32), dim3(256), 0, stream>>>(
      W_dt, Wdt_t, DMODEL, HEADS, DT_NPAD);
  f32_to_bf16_transpose_kernel<<<dim3(DMODEL / 32, INTER / 32), dim3(256), 0, stream>>>(
      W_out, Wout_t, INTER, DMODEL, DMODEL);

  // 3) gate = hs @ W_z        [4096,2048]x[2048,4096]
  gemm_bf16_wmma_kernel<<<dim3(INTER / 128, (unsigned)(M / 128)), dim3(256), 0, stream>>>(
      hs_bf, Wz_t, gate, (int)M, DMODEL, INTER);

  // 4) xBC_raw = hs @ W_xBC   [4096,2048]x[2048,4352]
  gemm_bf16_wmma_kernel<<<dim3(CONV_DIM / 128, (unsigned)(M / 128)), dim3(256), 0, stream>>>(
      hs_bf, WxBC_t, xBC_raw, (int)M, DMODEL, CONV_DIM);

  // 5) dt_raw = hs @ W_dt_pad [4096,2048]x[2048,128]
  gemm_bf16_wmma_kernel<<<dim3(DT_NPAD / 128, (unsigned)(M / 128)), dim3(256), 0, stream>>>(
      hs_bf, Wdt_t, dt_raw, (int)M, DMODEL, DT_NPAD);

  // 6) dt = softplus(dt_raw + dt_bias), compact to [M,64]
  dt_act_kernel<<<dim3((unsigned)((n_dt + 255) / 256)), dim3(256), 0, stream>>>(
      dt_raw, dt_bias, dtb, n_dt);

  // 7) causal depthwise conv + SiLU
  conv_silu_kernel<<<dim3((unsigned)((n_xBC + 255) / 256)), dim3(256), 0, stream>>>(
      xBC_raw, conv_w, conv_b, xBC_act, n_xBC);

  // 8) sequential SSM scan, parallel over (head, batch)
  ssm_scan_kernel<<<dim3(HEADS, B_SZ), dim3(256), 0, stream>>>(
      xBC_act, dtb, A_log, Dvec, ybuf);

  // 9) gated RMSNorm -> bf16
  rmsnorm_gate_kernel<<<dim3((unsigned)M), dim3(256), 0, stream>>>(
      ybuf, gate, norm_w, normed_bf);

  // 10) out = normed @ W_out  [4096,4096]x[4096,2048] -> f32 d_out
  gemm_bf16_wmma_kernel<<<dim3(DMODEL / 128, (unsigned)(M / 128)), dim3(256), 0, stream>>>(
      normed_bf, Wout_t, out, (int)M, INTER, DMODEL);
}
